// SGATHA_54262616817873
// MI455X (gfx1250) — matrix operand order repaired and checked by reading, not compiled, not run on hardware
//
#include <hip/hip_runtime.h>
#include <hip/hip_bf16.h>

#define N_NODES 50000
#define N_EDGES 800000
#define H_HEADS 2
#define IN_DIM  128
#define HID     64
#define NCLS    40
#define NEG_SLOPE 0.2f
#define BN_EPS  1e-5f

typedef __attribute__((ext_vector_type(16))) _Float16 v16h;
typedef __attribute__((ext_vector_type(8)))  float    v8f;

// ---------- small utilities ----------

__global__ void fill_f32(float* p, float v, int n) {
    int i = blockIdx.x * blockDim.x + threadIdx.x;
    int stride = gridDim.x * blockDim.x;
    for (; i < n; i += stride) p[i] = v;
}

__device__ __forceinline__ unsigned f2k(float f) {
    unsigned u = __float_as_uint(f);
    return (u >> 31) ? ~u : (u | 0x80000000u);   // order-preserving key
}
__device__ __forceinline__ float k2f(unsigned k) {
    unsigned u = (k >> 31) ? (k ^ 0x80000000u) : ~k;
    return __uint_as_float(u);
}

__global__ void fill_key(unsigned* p, float v, int n) {
    unsigned k = f2k(v);
    int i = blockIdx.x * blockDim.x + threadIdx.x;
    int stride = gridDim.x * blockDim.x;
    for (; i < n; i += stride) p[i] = k;
}

// ---------- degrees ----------

__global__ void deg_accum(const int* __restrict__ src, const int* __restrict__ dst,
                          float* degO, float* degI) {
    int e = blockIdx.x * blockDim.x + threadIdx.x;
    if (e < N_EDGES) {
        atomicAdd(&degO[src[e]], 1.0f);
        atomicAdd(&degI[dst[e]], 1.0f);
    }
}

// oi <- out_deg^-1/2 ; isq <- in_deg^1/2 ; sio <- oi*isq  (deg accumulated in oi/isq slots)
__global__ void deg_final(float* oi, float* isq, float* sio) {
    int n = blockIdx.x * blockDim.x + threadIdx.x;
    if (n < N_NODES) {
        float o  = fmaxf(oi[n], 1.0f);
        float in = fmaxf(isq[n], 1.0f);
        float a = rsqrtf(o);
        float b = sqrtf(in);
        oi[n] = a; isq[n] = b; sio[n] = a * b;
    }
}

// ---------- WMMA GEMM with LDS staging ----------
// Y[M x Nc] = X[M x Kd] @ W[Kd x Nc]; fp32 in/out, f16 MACs, fp32 accumulate.
// Block = 128 threads (4 waves). Block computes a 16-row x 64-col slab:
// A tile (16x32) and B slab (32x64) are cooperatively loaded (coalesced,
// fully vectorized), converted to f16 and stored into LDS *already in WMMA
// fragment order*, so each lane fetches its v16h fragment with two
// ds_load_b128. Requires M%16==0, Kd%32==0, Nc%16==0.

__global__ void gemm_wmma(const float* __restrict__ X, const float* __restrict__ W,
                          float* __restrict__ Y, int Kd, int Nc) {
    __shared__ __attribute__((aligned(32))) _Float16 lds_a[16 * 32];       // 1 KB
    __shared__ __attribute__((aligned(32))) _Float16 lds_b[4][16 * 32];    // 4 KB

    const int t      = threadIdx.x;
    const int wave   = t >> 5;
    const int lane   = t & 31;
    const int laneLo = lane & 15;
    const int laneHi = lane >> 4;
    const int row0   = blockIdx.x * 16;
    const int colBase = blockIdx.y * 64;
    const int tileN  = blockIdx.y * 4 + wave;
    const bool active = (tileN * 16 < Nc);       // wave-uniform
    const int col0   = tileN * 16;

    // B loader: thread handles 16 contiguous, 16-aligned columns. Since
    // Nc % 16 == 0 the whole group is uniformly in or out of range.
    const int bk  = t >> 2;                      // 0..31
    const int bnc = (t & 3) * 16;                // 0,16,32,48
    const bool bok = (colBase + bnc) < Nc;

    v8f acc = {};
    for (int k0 = 0; k0 < Kd; k0 += 32) {
        // ---- cooperative A tile load: 16 rows x 32 k, 4 consecutive k per thread ----
        {
            int base = t * 4;                    // linear over row*32 + k
            int row  = base >> 5;
            int k    = base & 31;
            const float4 xv = *(const float4*)(X + (size_t)(row0 + row) * Kd + k0 + k);
            float xs[4] = {xv.x, xv.y, xv.z, xv.w};
#pragma unroll
            for (int j = 0; j < 4; ++j) {
                int kk   = k + j;
                // A fragment in this code's contiguous k order (B uses the same): laneHi = k>>4 ;
                // slot i = k&15 ; lane = row + 16*laneHi
                int lhi  = kk >> 4;
                int slot = kk & 15;
                lds_a[(row + 16 * lhi) * 16 + slot] = (_Float16)xs[j];
            }
        }
        // ---- cooperative B slab load: 32 k x 64 cols, 16 consecutive cols per thread ----
        {
            float vals[16];
            if (bok) {
                const float* wr = W + (size_t)(k0 + bk) * Nc + colBase + bnc;
                const float4 w0 = *(const float4*)(wr + 0);
                const float4 w1 = *(const float4*)(wr + 4);
                const float4 w2 = *(const float4*)(wr + 8);
                const float4 w3 = *(const float4*)(wr + 12);
                vals[0] = w0.x;  vals[1] = w0.y;  vals[2] = w0.z;  vals[3] = w0.w;
                vals[4] = w1.x;  vals[5] = w1.y;  vals[6] = w1.z;  vals[7] = w1.w;
                vals[8] = w2.x;  vals[9] = w2.y;  vals[10] = w2.z; vals[11] = w2.w;
                vals[12] = w3.x; vals[13] = w3.y; vals[14] = w3.z; vals[15] = w3.w;
            } else {
#pragma unroll
                for (int j = 0; j < 16; ++j) vals[j] = 0.0f;
            }
#pragma unroll
            for (int j = 0; j < 16; ++j) {
                // B fragment in this code's contiguous k order (same as A): lane = (n&15) + 16*(k>>4) ; slot = k&15
                int tile = (bnc + j) >> 4;
                int ln   = ((bnc + j) & 15) + 16 * (bk >> 4);
                lds_b[tile][ln * 16 + (bk & 15)] = (_Float16)vals[j];
            }
        }
        __syncthreads();

        if (active) {
            v16h a = *(const v16h*)(lds_a + lane * 16);
            v16h b = *(const v16h*)(lds_b[wave] + lane * 16);
            acc = __builtin_amdgcn_wmma_f32_16x16x32_f16(
                /*neg_a=*/false, a, /*neg_b=*/false, b,
                /*c_mod=*/(short)0, acc, /*reuse_a=*/false, /*reuse_b=*/false);
        }
        __syncthreads();
    }

    if (active) {
        // C/D layout: VGPR r -> M = laneHi*8 + r, N = laneLo
#pragma unroll
        for (int r = 0; r < 8; ++r)
            Y[(size_t)(row0 + laneHi * 8 + r) * Nc + col0 + laneLo] = acc[r];
    }
}

// ---------- attention ----------

// out[n*H+h] = sum_f Hbuf[n, h*F+f] * attn[h*F+f]
__global__ void attn_scores(const float* __restrict__ Hbuf, const float* __restrict__ attn,
                            float* __restrict__ out, int F) {
    int i = blockIdx.x * blockDim.x + threadIdx.x;
    if (i >= N_NODES * H_HEADS) return;
    int n = i >> 1, h = i & 1;
    const float* hp = Hbuf + (size_t)n * (H_HEADS * F) + h * F;
    const float* ap = attn + h * F;
    float s = 0.0f;
    for (int f = 0; f < F; ++f) s += hp[f] * ap[f];
    out[i] = s;
}

__device__ __forceinline__ float edge_logit(const int* src, const int* dst,
                                            const float* el, const float* er,
                                            int e, int h) {
    float v = el[src[e] * H_HEADS + h] + er[dst[e] * H_HEADS + h];
    return (v >= 0.0f) ? v : NEG_SLOPE * v;
}

__global__ void edge_max(const int* __restrict__ src, const int* __restrict__ dst,
                         const float* __restrict__ el, const float* __restrict__ er,
                         unsigned* __restrict__ emax) {
    int i = blockIdx.x * blockDim.x + threadIdx.x;
    if (i >= N_EDGES * H_HEADS) return;
    int e = i >> 1, h = i & 1;
    float v = edge_logit(src, dst, el, er, e, h);
    atomicMax(&emax[dst[e] * H_HEADS + h], f2k(v));
}

__global__ void edge_exp(const int* __restrict__ src, const int* __restrict__ dst,
                         const float* __restrict__ el, const float* __restrict__ er,
                         const unsigned* __restrict__ emax,
                         float* __restrict__ alpha, float* __restrict__ esum) {
    int i = blockIdx.x * blockDim.x + threadIdx.x;
    if (i >= N_EDGES * H_HEADS) return;
    int e = i >> 1, h = i & 1;
    float v  = edge_logit(src, dst, el, er, e, h);
    float ex = expf(v - k2f(emax[dst[e] * H_HEADS + h]));
    alpha[i] = ex;
    atomicAdd(&esum[dst[e] * H_HEADS + h], ex);
}

__global__ void edge_div(const int* __restrict__ dst, float* __restrict__ alpha,
                         const float* __restrict__ esum) {
    int i = blockIdx.x * blockDim.x + threadIdx.x;
    if (i >= N_EDGES * H_HEADS) return;
    int e = i >> 1, h = i & 1;
    alpha[i] /= esum[dst[e] * H_HEADS + h];
}

// ---------- propagation: next[dst, f] += prev[src, f] * scale[src] * alpha[e, head(f)] ----------

__global__ void propagate(const int* __restrict__ src, const int* __restrict__ dst,
                          const float* __restrict__ alpha, const float* __restrict__ scale,
                          const float* __restrict__ prev, float* __restrict__ next,
                          int Ftot, int Fh, int total) {
    int i = blockIdx.x * blockDim.x + threadIdx.x;
    if (i >= total) return;
    int fq = Ftot >> 2;
    int e  = i / fq;
    int f  = (i - e * fq) * 4;
    int s = src[e], d = dst[e];
    int head = f / Fh;                       // Fh % 4 == 0, so group stays in one head
    float w = alpha[e * H_HEADS + head] * scale[s];
    const float4 v = *(const float4*)(prev + (size_t)s * Ftot + f);
    float* np = next + (size_t)d * Ftot + f;
    atomicAdd(np + 0, v.x * w);
    atomicAdd(np + 1, v.y * w);
    atomicAdd(np + 2, v.z * w);
    atomicAdd(np + 3, v.w * w);
}

// out[n,f] = Q[n,f]*isq[n] + R[n,f]
__global__ void combine_res(const float* __restrict__ Q, const float* __restrict__ R,
                            const float* __restrict__ isq, float* __restrict__ out, int Ftot) {
    int i = blockIdx.x * blockDim.x + threadIdx.x;
    if (i >= N_NODES * Ftot) return;
    int n = i / Ftot;
    out[i] = Q[i] * isq[n] + R[i];
}

// ---------- batch norm ----------

__global__ void stats_kernel(const float* __restrict__ X, float* mu, float* ss) {
    int c = blockIdx.x;                 // 128 channels
    float s = 0.0f, s2 = 0.0f;
    for (int n = threadIdx.x; n < N_NODES; n += blockDim.x) {
        float v = X[(size_t)n * (H_HEADS * HID) + c];
        s += v; s2 += v * v;
    }
    atomicAdd(&mu[c], s);
    atomicAdd(&ss[c], s2);
}

__global__ void bn_relu(float* __restrict__ X, const float* __restrict__ mu,
                        const float* __restrict__ ss, const float* __restrict__ gamma,
                        const float* __restrict__ beta) {
    int i = blockIdx.x * blockDim.x + threadIdx.x;
    if (i >= N_NODES * H_HEADS * HID) return;
    int c = i & (H_HEADS * HID - 1);    // 128 channels
    float mean = mu[c] * (1.0f / N_NODES);
    float var  = ss[c] * (1.0f / N_NODES) - mean * mean;
    float v = (X[i] - mean) * rsqrtf(var + BN_EPS) * gamma[c] + beta[c];
    X[i] = fmaxf(v, 0.0f);
}

// ---------- final head-mean + bias ----------

__global__ void final_out(const float* __restrict__ C, const float* __restrict__ R,
                          const float* __restrict__ isq, const float* __restrict__ bias,
                          float* __restrict__ out) {
    int i = blockIdx.x * blockDim.x + threadIdx.x;
    if (i >= N_NODES * NCLS) return;
    int n = i / NCLS, c = i - n * NCLS;
    float a = C[(size_t)n * (H_HEADS * NCLS) + c]        * isq[n] + R[(size_t)n * (H_HEADS * NCLS) + c];
    float b = C[(size_t)n * (H_HEADS * NCLS) + NCLS + c] * isq[n] + R[(size_t)n * (H_HEADS * NCLS) + NCLS + c];
    out[i] = 0.5f * (a + b) + bias[c];
}

// ---------- launcher ----------

static inline int cdiv(long long a, int b) { return (int)((a + b - 1) / b); }

extern "C" void kernel_launch(void* const* d_in, const int* in_sizes, int n_in,
                              void* d_out, int out_size, void* d_ws, size_t ws_size,
                              hipStream_t stream) {
    (void)in_sizes; (void)n_in; (void)out_size; (void)ws_size;
    const float* feat    = (const float*)d_in[0];
    const int*   src     = (const int*)  d_in[1];
    const int*   dst     = (const int*)  d_in[2];
    const float* fc_w0   = (const float*)d_in[3];
    const float* attn_l0 = (const float*)d_in[4];
    const float* attn_r0 = (const float*)d_in[5];
    const float* res_w0  = (const float*)d_in[6];
    const float* fc_w1   = (const float*)d_in[7];
    const float* attn_l1 = (const float*)d_in[8];
    const float* attn_r1 = (const float*)d_in[9];
    const float* res_w1  = (const float*)d_in[10];
    const float* bn_g    = (const float*)d_in[11];
    const float* bn_b    = (const float*)d_in[12];
    const float* bias    = (const float*)d_in[13];
    float* out = (float*)d_out;

    // workspace layout (floats, 16-elt aligned)
    float* ws = (float*)d_ws;
    size_t o = 0;
    auto alloc = [&](size_t n) { size_t r = o; o += (n + 15) & ~(size_t)15; return ws + r; };
    float*    oi    = alloc(N_NODES);
    float*    isq   = alloc(N_NODES);
    float*    sio   = alloc(N_NODES);
    float*    el    = alloc((size_t)N_NODES * H_HEADS);
    float*    er    = alloc((size_t)N_NODES * H_HEADS);
    unsigned* emax  = (unsigned*)alloc((size_t)N_NODES * H_HEADS);
    float*    esum  = alloc((size_t)N_NODES * H_HEADS);
    float*    alpha = alloc((size_t)N_EDGES * H_HEADS);
    float*    bufA  = alloc((size_t)N_NODES * 128);
    float*    bufB  = alloc((size_t)N_NODES * 128);
    float*    bufC  = alloc((size_t)N_NODES * 128);
    float*    bufR  = alloc((size_t)N_NODES * 128);
    float*    mu    = alloc(128);
    float*    ss    = alloc(128);

    const int T = 256;
    const dim3 fillG(1024), fillB(T);
    const int NH = N_NODES * H_HEADS;
    const int EH = N_EDGES * H_HEADS;

    // ---- degrees ----
    fill_f32<<<fillG, fillB, 0, stream>>>(oi, 0.f, N_NODES);
    fill_f32<<<fillG, fillB, 0, stream>>>(isq, 0.f, N_NODES);
    deg_accum<<<cdiv(N_EDGES, T), T, 0, stream>>>(src, dst, oi, isq);
    deg_final<<<cdiv(N_NODES, T), T, 0, stream>>>(oi, isq, sio);

    // ================= layer 0 (F = 64, Ftot = 128) =================
    gemm_wmma<<<dim3(N_NODES / 16, 2), 128, 0, stream>>>(feat, fc_w0, bufA, IN_DIM, H_HEADS * HID);
    gemm_wmma<<<dim3(N_NODES / 16, 2), 128, 0, stream>>>(feat, res_w0, bufR, IN_DIM, H_HEADS * HID);
    attn_scores<<<cdiv(NH, T), T, 0, stream>>>(bufA, attn_l0, el, HID);
    attn_scores<<<cdiv(NH, T), T, 0, stream>>>(bufA, attn_r0, er, HID);

    fill_key<<<fillG, fillB, 0, stream>>>(emax, -3.0e38f, NH);
    edge_max<<<cdiv(EH, T), T, 0, stream>>>(src, dst, el, er, emax);
    fill_f32<<<fillG, fillB, 0, stream>>>(esum, 0.f, NH);
    edge_exp<<<cdiv(EH, T), T, 0, stream>>>(src, dst, el, er, emax, alpha, esum);
    edge_div<<<cdiv(EH, T), T, 0, stream>>>(dst, alpha, esum);

    {
        const int Ftot = 128, Fh = HID;
        const int total = N_EDGES * (Ftot / 4);
        fill_f32<<<fillG, fillB, 0, stream>>>(bufB, 0.f, N_NODES * Ftot);
        propagate<<<cdiv(total, T), T, 0, stream>>>(src, dst, alpha, oi,  bufA, bufB, Ftot, Fh, total);
        fill_f32<<<fillG, fillB, 0, stream>>>(bufA, 0.f, N_NODES * Ftot);
        propagate<<<cdiv(total, T), T, 0, stream>>>(src, dst, alpha, sio, bufB, bufA, Ftot, Fh, total);
        fill_f32<<<fillG, fillB, 0, stream>>>(bufB, 0.f, N_NODES * Ftot);
        propagate<<<cdiv(total, T), T, 0, stream>>>(src, dst, alpha, sio, bufA, bufB, Ftot, Fh, total);
        combine_res<<<cdiv(N_NODES * Ftot, T), T, 0, stream>>>(bufB, bufR, isq, bufA, Ftot);
    }

    // batch norm (training-mode batch stats) + ReLU, in place on bufA
    fill_f32<<<1, 256, 0, stream>>>(mu, 0.f, 128);
    fill_f32<<<1, 256, 0, stream>>>(ss, 0.f, 128);
    stats_kernel<<<128, 256, 0, stream>>>(bufA, mu, ss);
    bn_relu<<<cdiv(N_NODES * 128, T), T, 0, stream>>>(bufA, mu, ss, bn_g, bn_b);

    // ================= layer 1 (F = 40, Ftot = 80) =================
    gemm_wmma<<<dim3(N_NODES / 16, 2), 128, 0, stream>>>(bufA, fc_w1, bufB, H_HEADS * HID, H_HEADS * NCLS);
    gemm_wmma<<<dim3(N_NODES / 16, 2), 128, 0, stream>>>(bufA, res_w1, bufR, H_HEADS * HID, H_HEADS * NCLS);
    attn_scores<<<cdiv(NH, T), T, 0, stream>>>(bufB, attn_l1, el, NCLS);
    attn_scores<<<cdiv(NH, T), T, 0, stream>>>(bufB, attn_r1, er, NCLS);

    fill_key<<<fillG, fillB, 0, stream>>>(emax, -3.0e38f, NH);
    edge_max<<<cdiv(EH, T), T, 0, stream>>>(src, dst, el, er, emax);
    fill_f32<<<fillG, fillB, 0, stream>>>(esum, 0.f, NH);
    edge_exp<<<cdiv(EH, T), T, 0, stream>>>(src, dst, el, er, emax, alpha, esum);
    edge_div<<<cdiv(EH, T), T, 0, stream>>>(dst, alpha, esum);

    {
        const int Ftot = 80, Fh = NCLS;
        const int total = N_EDGES * (Ftot / 4);
        fill_f32<<<fillG, fillB, 0, stream>>>(bufC, 0.f, N_NODES * Ftot);
        propagate<<<cdiv(total, T), T, 0, stream>>>(src, dst, alpha, oi,  bufB, bufC, Ftot, Fh, total);
        fill_f32<<<fillG, fillB, 0, stream>>>(bufB, 0.f, N_NODES * Ftot);
        propagate<<<cdiv(total, T), T, 0, stream>>>(src, dst, alpha, sio, bufC, bufB, Ftot, Fh, total);
        fill_f32<<<fillG, fillB, 0, stream>>>(bufC, 0.f, N_NODES * Ftot);
        propagate<<<cdiv(total, T), T, 0, stream>>>(src, dst, alpha, sio, bufB, bufC, Ftot, Fh, total);
    }

    final_out<<<cdiv(N_NODES * NCLS, T), T, 0, stream>>>(bufC, bufR, isq, bias, out);
}
